// EF_expansion_59880434041529
// MI455X (gfx1250) — compile-verified, matrix operand order checked by reading, not run on hardware
//
#include <hip/hip_runtime.h>
#include <hip/hip_bf16.h>

// EdgeConv block for MI455X (gfx1250, wave32).
// KNN Gram via V_WMMA_F32_16X16X4_F32 (f32, cancellation-sensitive),
// MLP layers via V_WMMA_F32_16X16X32_BF16 (throughput path).
// ALL gemm operands (weights + intermediates F/E/E2) stored fragment-native:
// [chunk][jt][lane][8 dwords] -> fragment load = 2 x global_load_b128,
// layer output store = 1 x global_store_b128 per row-tile.

#define NB   16          // batches
#define CH   64          // channels C
#define NP   2048        // points N
#define KNN  4           // neighbors
#define OUTC 64
#define NKC  (NP*KNN)    // 8192 gemm columns per batch
#define NJT  (NKC/16)    // 512 column tiles per batch
#define FRAGB (NJT*256)  // dwords per chunk per batch (NJT*32*8)
#define NEGINF (-3.4e38f)

typedef __attribute__((ext_vector_type(2)))  float        v2f;
typedef __attribute__((ext_vector_type(8)))  float        v8f;
typedef __attribute__((ext_vector_type(16))) __bf16       v16bf;
typedef __attribute__((ext_vector_type(8)))  unsigned int v8u;

__device__ __forceinline__ unsigned short f2bf(float f) {
    unsigned u = __builtin_bit_cast(unsigned, f);
    u += 0x7FFFu + ((u >> 16) & 1u);           // round-to-nearest-even
    return (unsigned short)(u >> 16);
}
__device__ __forceinline__ unsigned pack2bf(float lo, float hi) {
    return (unsigned)f2bf(lo) | ((unsigned)f2bf(hi) << 16);
}
__device__ __forceinline__ unsigned relu_pk(unsigned u) {   // relu both bf16 halves
    unsigned lo = (u & 0x8000u)     ? 0u : (u & 0xFFFFu);
    unsigned hw = (u & 0x80000000u) ? 0u : (u & 0xFFFF0000u);
    return lo | hw;
}

// fragment-native addressing: dword offset of lane's 8-dword group
__device__ __forceinline__ size_t fragbase(int kc, int jt, int lane) {
    return ((size_t)(kc * NJT + jt) * 32 + lane) * 8;
}
// 32B fragment load as 2 x b128
__device__ __forceinline__ v16bf load_frag8(const unsigned* __restrict__ P, size_t base) {
    const uint4* p = (const uint4*)(P + base);
    uint4 a = p[0], b = p[1];
    v8u u;
    u[0] = a.x; u[1] = a.y; u[2] = a.z; u[3] = a.w;
    u[4] = b.x; u[5] = b.y; u[6] = b.z; u[7] = b.w;
    return __builtin_bit_cast(v16bf, u);
}
__device__ __forceinline__ v16bf load_afrag(const unsigned* __restrict__ Wf, int NKcW,
                                            int rt, int kc, int lane) {
    return load_frag8(Wf, ((size_t)(rt * NKcW + kc) * 32 + lane) * 8);
}

// ---------------------------------------------------------------- xx[b][n]
__global__ void k_xx(const float* __restrict__ x, float* __restrict__ xx) {
    int t = blockIdx.x * blockDim.x + threadIdx.x;   // over NB*NP
    int b = t / NP, n = t % NP;
    const float* xp = x + (size_t)b * CH * NP + n;
    float s = 0.f;
#pragma unroll 8
    for (int c = 0; c < CH; ++c) { float v = xp[(size_t)c * NP]; s += v * v; }
    xx[t] = s;
}

// ------------------------------------------------- KNN: f32 WMMA + top-4
__global__ void __launch_bounds__(32)
k_knn(const float* __restrict__ x, const float* __restrict__ xx,
      int* __restrict__ idxout) {
    __shared__ float tileLds[16 * 16];

    const int b  = blockIdx.y;
    const int rt = blockIdx.x;          // row tile (0..127)
    const int l  = threadIdx.x;         // 0..31
    const int lm = l & 15;
    const int hi = l >> 4;              // 0/1 half

    const float* xb = x + (size_t)b * CH * NP;
    const int m = rt * 16 + lm;         // my A row (point index)
    const float xm = xx[b * NP + m];

    v2f a[16];
#pragma unroll
    for (int kc = 0; kc < 16; ++kc) {
        int c0 = kc * 4 + hi * 2;
        a[kc].x = xb[(size_t)c0 * NP + m];
        a[kc].y = xb[(size_t)(c0 + 1) * NP + m];
    }

    float t0 = NEGINF, t1 = NEGINF, t2 = NEGINF, t3 = NEGINF;
    int   i0 = 0, i1 = 0, i2 = 0, i3 = 0;

    for (int ct = 0; ct < NP / 16; ++ct) {
        const int ncol = ct * 16 + lm;
        v8f acc = {};
#pragma unroll
        for (int kc = 0; kc < 16; ++kc) {
            int c0 = kc * 4 + hi * 2;
            v2f bb;
            bb.x = xb[(size_t)c0 * NP + ncol];
            bb.y = xb[(size_t)(c0 + 1) * NP + ncol];
            acc = __builtin_amdgcn_wmma_f32_16x16x4_f32(
                false, a[kc], false, bb, (short)0, acc, false, false);
        }
        __syncthreads();
#pragma unroll
        for (int v = 0; v < 8; ++v) tileLds[(v + hi * 8) * 16 + lm] = acc[v];
        __syncthreads();
#pragma unroll
        for (int i = 0; i < 8; ++i) {
            int col = ct * 16 + hi * 8 + i;
            float nd = 2.f * tileLds[lm * 16 + hi * 8 + i] - xm - xx[b * NP + col];
            bool g0 = nd > t0, g1 = nd > t1, g2 = nd > t2, g3 = nd > t3;
            float u3 = g2 ? t2 : (g3 ? nd : t3);  int j3 = g2 ? i2 : (g3 ? col : i3);
            float u2 = g1 ? t1 : (g2 ? nd : t2);  int j2 = g1 ? i1 : (g2 ? col : i2);
            float u1 = g0 ? t0 : (g1 ? nd : t1);  int j1 = g0 ? i0 : (g1 ? col : i1);
            t0 = g0 ? nd : t0;                    i0 = g0 ? col : i0;
            t1 = u1; i1 = j1; t2 = u2; i2 = j2; t3 = u3; i3 = j3;
        }
    }

    float c0 = __shfl(t0, l ^ 16, 32), c1 = __shfl(t1, l ^ 16, 32);
    float c2 = __shfl(t2, l ^ 16, 32), c3 = __shfl(t3, l ^ 16, 32);
    int   C0 = __shfl(i0, l ^ 16, 32), C1 = __shfl(i1, l ^ 16, 32);
    int   C2 = __shfl(i2, l ^ 16, 32), C3 = __shfl(i3, l ^ 16, 32);

    if (hi == 0) {
        int obase = (b * NP + m) * KNN;
#pragma unroll
        for (int q = 0; q < 4; ++q) {
            bool tA = (t0 > c0) || (t0 == c0 && i0 < C0);
            idxout[obase + q] = tA ? i0 : C0;
            float a0n = tA ? t1 : t0;  int A0n = tA ? i1 : i0;
            float a1n = tA ? t2 : t1;  int A1n = tA ? i2 : i1;
            float a2n = tA ? t3 : t2;  int A2n = tA ? i3 : i2;
            float a3n = tA ? NEGINF : t3;
            float c0n = tA ? c0 : c1;  int C0n = tA ? C0 : C1;
            float c1n = tA ? c1 : c2;  int C1n = tA ? C1 : C2;
            float c2n = tA ? c2 : c3;  int C2n = tA ? C2 : C3;
            float c3n = tA ? c3 : NEGINF;
            t0 = a0n; i0 = A0n; t1 = a1n; i1 = A1n; t2 = a2n; i2 = A2n; t3 = a3n;
            c0 = c0n; C0 = C0n; c1 = c1n; C1 = C1n; c2 = c2n; C2 = C2n; c3 = c3n;
        }
    }
}

// ----------- pack weights f32 -> bf16 in WMMA A-fragment order (32B/lane)
__device__ __forceinline__ void packfrag(const float* __restrict__ W, int Cin,
                                         int NKc, int t, unsigned* __restrict__ out) {
    int v = t & 7, lane = (t >> 3) & 31, f = t >> 8;
    int rt = f / NKc, kc = f % NKc;
    int lm = lane & 15, hi = lane >> 4;
    int row = rt * 16 + lm;
    int cp = kc * 16 + (v & 3) + ((v >> 2) << 3) + hi * 4;   // bf16 pair column
    out[t] = pack2bf(W[row * Cin + 2 * cp], W[row * Cin + 2 * cp + 1]);
}

__global__ void k_packw(const float* __restrict__ w1, const float* __restrict__ w2,
                        const float* __restrict__ w3,
                        unsigned* __restrict__ w1f, unsigned* __restrict__ w2f,
                        unsigned* __restrict__ w3f) {
    int t = blockIdx.x * blockDim.x + threadIdx.x;   // 0..12287 covers w2
    if (t < 48 * 256) packfrag(w2, 192, 6, t, w2f);  // 8 rt x 6 kc
    if (t < 16 * 256) packfrag(w1, 128, 4, t, w1f);  // 4 rt x 4 kc
    if (t <  8 * 256) packfrag(w3,  64, 2, t, w3f);  // 4 rt x 2 kc
}

// ------------- gather: feat -> F fragment-native (4 chunks), b128 stores
__global__ void k_gather(const float* __restrict__ x, const int* __restrict__ idx,
                         unsigned* __restrict__ F) {
    int t = blockIdx.x * blockDim.x + threadIdx.x;   // over NB*NKC
    int b = t / NKC, j = t % NKC;
    int n = j >> 2, k = j & 3;
    int nb = idx[(b * NP + n) * KNN + k];
    const float* xc = x + (size_t)b * CH * NP + n;
    const float* xn = x + (size_t)b * CH * NP + nb;
    unsigned* Fb = F + (size_t)b * 4 * FRAGB;
    const int jt = j >> 4, lm = j & 15;

#pragma unroll
    for (int kc = 0; kc < 4; ++kc) {                 // chunks 0,1=center 2,3=neighbor
        const float* src = (kc < 2) ? xc : xn;
        int pb0 = 16 * kc - ((kc < 2) ? 0 : 32);     // pair offset within src
#pragma unroll
        for (int h = 0; h < 2; ++h) {                // dest half-lane
            unsigned d[8];
#pragma unroll
            for (int v = 0; v < 8; ++v) {
                int cp = pb0 + 4 * h + (v & 3) + ((v >> 2) << 3);
                d[v] = pack2bf(src[(size_t)(2 * cp) * NP], src[(size_t)(2 * cp + 1) * NP]);
            }
            uint4* dst = (uint4*)(Fb + fragbase(kc, jt, lm + 16 * h));
            uint4 s0; s0.x = d[0]; s0.y = d[1]; s0.z = d[2]; s0.w = d[3];
            uint4 s1; s1.x = d[4]; s1.y = d[5]; s1.z = d[6]; s1.w = d[7];
            dst[0] = s0; dst[1] = s1;
        }
    }
}

// -------------------- layer1: E = relu(concat(w1@F + b1, F))  (6 chunks)
__global__ void __launch_bounds__(256)
k_layer1(const unsigned* __restrict__ F, const unsigned* __restrict__ w1f,
         const float* __restrict__ b1, unsigned* __restrict__ E) {
    const int b  = blockIdx.y;
    const int jt = blockIdx.x * 8 + (threadIdx.x >> 5);  // 512 col tiles
    const int l  = threadIdx.x & 31, hi = l >> 4;
    const unsigned* Fb = F + (size_t)b * 4 * FRAGB;
    unsigned* Eb = E + (size_t)b * 6 * FRAGB;

    v16bf bf[4];
#pragma unroll
    for (int kc = 0; kc < 4; ++kc) bf[kc] = load_frag8(Fb, fragbase(kc, jt, l));

#pragma unroll
    for (int rt = 0; rt < 4; ++rt) {
        v8f acc = {};
#pragma unroll
        for (int kc = 0; kc < 4; ++kc) {
            v16bf af = load_afrag(w1f, 4, rt, kc, l);
            acc = __builtin_amdgcn_wmma_f32_16x16x32_bf16(
                false, af, false, bf[kc], (short)0, acc, false, false);
        }
        const int rbase = rt * 16 + hi * 8;          // D rows rbase..rbase+7
        uint4 s;                                      // 4 pair-dwords, v = w + 4*(rt&1)
        s.x = pack2bf(fmaxf(acc[0] + b1[rbase + 0], 0.f), fmaxf(acc[1] + b1[rbase + 1], 0.f));
        s.y = pack2bf(fmaxf(acc[2] + b1[rbase + 2], 0.f), fmaxf(acc[3] + b1[rbase + 3], 0.f));
        s.z = pack2bf(fmaxf(acc[4] + b1[rbase + 4], 0.f), fmaxf(acc[5] + b1[rbase + 5], 0.f));
        s.w = pack2bf(fmaxf(acc[6] + b1[rbase + 6], 0.f), fmaxf(acc[7] + b1[rbase + 7], 0.f));
        *(uint4*)(Eb + fragbase(rt >> 1, jt, l) + 4 * (rt & 1)) = s;
    }
    // passthrough: E chunks 2..5 = relu(F chunks 0..3), reuse loaded fragments
#pragma unroll
    for (int kc = 0; kc < 4; ++kc) {
        v8u u = __builtin_bit_cast(v8u, bf[kc]);
        uint4 s0; s0.x = relu_pk(u[0]); s0.y = relu_pk(u[1]); s0.z = relu_pk(u[2]); s0.w = relu_pk(u[3]);
        uint4 s1; s1.x = relu_pk(u[4]); s1.y = relu_pk(u[5]); s1.z = relu_pk(u[6]); s1.w = relu_pk(u[7]);
        uint4* dst = (uint4*)(Eb + fragbase(kc + 2, jt, l));
        dst[0] = s0; dst[1] = s1;
    }
}

// ------------------------------ layer2: E2 = relu(w2@E + b2)  (4 chunks)
__global__ void __launch_bounds__(256)
k_layer2(const unsigned* __restrict__ E, const unsigned* __restrict__ w2f,
         const float* __restrict__ b2, unsigned* __restrict__ E2) {
    const int b  = blockIdx.y;
    const int jt = blockIdx.x * 8 + (threadIdx.x >> 5);
    const int l  = threadIdx.x & 31, hi = l >> 4;
    const unsigned* Eb = E + (size_t)b * 6 * FRAGB;
    unsigned* E2b = E2 + (size_t)b * 4 * FRAGB;

    v16bf bf[6];
#pragma unroll
    for (int kc = 0; kc < 6; ++kc) bf[kc] = load_frag8(Eb, fragbase(kc, jt, l));

#pragma unroll
    for (int rt = 0; rt < 8; ++rt) {
        v8f acc = {};
#pragma unroll
        for (int kc = 0; kc < 6; ++kc) {
            v16bf af = load_afrag(w2f, 6, rt, kc, l);
            acc = __builtin_amdgcn_wmma_f32_16x16x32_bf16(
                false, af, false, bf[kc], (short)0, acc, false, false);
        }
        const int rbase = rt * 16 + hi * 8;
        uint4 s;
        s.x = pack2bf(fmaxf(acc[0] + b2[rbase + 0], 0.f), fmaxf(acc[1] + b2[rbase + 1], 0.f));
        s.y = pack2bf(fmaxf(acc[2] + b2[rbase + 2], 0.f), fmaxf(acc[3] + b2[rbase + 3], 0.f));
        s.z = pack2bf(fmaxf(acc[4] + b2[rbase + 4], 0.f), fmaxf(acc[5] + b2[rbase + 5], 0.f));
        s.w = pack2bf(fmaxf(acc[6] + b2[rbase + 6], 0.f), fmaxf(acc[7] + b2[rbase + 7], 0.f));
        *(uint4*)(E2b + fragbase(rt >> 1, jt, l) + 4 * (rt & 1)) = s;
    }
}

// -------- layer3: e3 = w3@e2 + b3, max over K, out[b][o][2n+r]  (f32 out)
__global__ void __launch_bounds__(256)
k_layer3(const unsigned* __restrict__ E2, const unsigned* __restrict__ w3f,
         const float* __restrict__ b3, float* __restrict__ out) {
    const int b  = blockIdx.y;
    const int jt = blockIdx.x * 8 + (threadIdx.x >> 5);
    const int l  = threadIdx.x & 31, lm = l & 15, hi = l >> 4;
    const unsigned* E2b = E2 + (size_t)b * 4 * FRAGB;
    const int col = jt * 16 + lm;     // j = n*4 + k ; lanes in groups of 4 share n

#pragma unroll
    for (int r = 0; r < 2; ++r) {     // e2 channel halves -> output m = 2n + r
        v16bf bfr[2];
#pragma unroll
        for (int kc = 0; kc < 2; ++kc)
            bfr[kc] = load_frag8(E2b, fragbase(r * 2 + kc, jt, l));
#pragma unroll
        for (int rt = 0; rt < 4; ++rt) {
            v8f acc = {};
#pragma unroll
            for (int kc = 0; kc < 2; ++kc) {
                v16bf af = load_afrag(w3f, 2, rt, kc, l);
                acc = __builtin_amdgcn_wmma_f32_16x16x32_bf16(
                    false, af, false, bfr[kc], (short)0, acc, false, false);
            }
#pragma unroll
            for (int v = 0; v < 8; ++v) {
                int o = rt * 16 + hi * 8 + v;
                float e3 = acc[v] + b3[o];
                e3 = fmaxf(e3, __shfl_xor(e3, 1, 32));   // max over K=4
                e3 = fmaxf(e3, __shfl_xor(e3, 2, 32));
                if ((lm & 3) == 0) {
                    int n = col >> 2;
                    out[((size_t)b * OUTC + o) * (2 * NP) + 2 * n + r] = e3;
                }
            }
        }
    }
}

// ---------------------------------------------------------------- launcher
extern "C" void kernel_launch(void* const* d_in, const int* in_sizes, int n_in,
                              void* d_out, int out_size, void* d_ws, size_t ws_size,
                              hipStream_t stream) {
    const float* x  = (const float*)d_in[0];
    const float* w1 = (const float*)d_in[1];
    const float* b1 = (const float*)d_in[2];
    const float* w2 = (const float*)d_in[3];
    const float* b2 = (const float*)d_in[4];
    const float* w3 = (const float*)d_in[5];
    const float* b3 = (const float*)d_in[6];
    float* out = (float*)d_out;

    char* ws = (char*)d_ws;
    size_t off = 0;
    float*    xx  = (float*)(ws + off);    off += (size_t)NB * NP * 4;              // 128 KB
    int*      idx = (int*)(ws + off);      off += (size_t)NB * NP * KNN * 4;        // 512 KB
    unsigned* w1f = (unsigned*)(ws + off); off += (size_t)16 * 256 * 4;             // 16 KB
    unsigned* w2f = (unsigned*)(ws + off); off += (size_t)48 * 256 * 4;             // 48 KB
    unsigned* w3f = (unsigned*)(ws + off); off += (size_t)8  * 256 * 4;             // 8 KB
    unsigned* F   = (unsigned*)(ws + off); off += (size_t)NB * 4 * FRAGB * 4;       // 32 MB
    unsigned* E   = (unsigned*)(ws + off); off += (size_t)NB * 6 * FRAGB * 4;       // 48 MB
    unsigned* E2  = (unsigned*)(ws + off); off += (size_t)NB * 4 * FRAGB * 4;       // 32 MB

    k_xx    <<<dim3((NB * NP) / 256), 256, 0, stream>>>(x, xx);
    k_knn   <<<dim3(NP / 16, NB), 32, 0, stream>>>(x, xx, idx);
    k_packw <<<dim3(48), 256, 0, stream>>>(w1, w2, w3, w1f, w2f, w3f);
    k_gather<<<dim3((NB * NKC) / 256), 256, 0, stream>>>(x, idx, F);
    k_layer1<<<dim3(64, NB), 256, 0, stream>>>(F, w1f, b1, E);
    k_layer2<<<dim3(64, NB), 256, 0, stream>>>(E, w2f, b2, E2);
    k_layer3<<<dim3(64, NB), 256, 0, stream>>>(E2, w3f, b3, out);
}
